// MyEGNNNet_40991167873102
// MI455X (gfx1250) — compile-verified, matrix operand order checked by reading, not run on hardware
//
#include <hip/hip_runtime.h>
#include <hip/hip_bf16.h>
#include <math.h>

// ---------------------------------------------------------------------------
// Problem constants (from the reference)
// ---------------------------------------------------------------------------
#define BB   4
#define N0_  50000
#define N1_  25000
#define N2_  12500
#define E0_  800000
#define E1_  400000
#define CIN_ 32
#define CS_  16
#define COUT_ 32

typedef __attribute__((ext_vector_type(16))) _Float16 v16h;
typedef __attribute__((ext_vector_type(8)))  _Float16 v8h;
typedef __attribute__((ext_vector_type(4)))  _Float16 v4h;
typedef __attribute__((ext_vector_type(8)))  float    v8f;

// small-params layout inside ws (floats)
#define SML_QV1 0
#define SML_KV1 16
#define SML_S31 32
#define SML_QV2 48
#define SML_KV2 80
#define SML_S32 112
#define SML_SIZE 128

// ---------------------------------------------------------------------------
// Precompute qv = Q @ Watt[0:C], kv = K @ Watt[C:2C], s3 = We . Watt[2C:3C]
// (collapses the per-edge 16x16 / 32x32 matmuls into dot products)
// ---------------------------------------------------------------------------
__global__ void prep_kernel(const float* __restrict__ Q1, const float* __restrict__ K1,
                            const float* __restrict__ Watt1, const float* __restrict__ We1,
                            const float* __restrict__ Q2, const float* __restrict__ K2,
                            const float* __restrict__ Watt2, const float* __restrict__ We2,
                            float* __restrict__ sml) {
    int c = threadIdx.x;  // 64 threads
    if (c < CS_) {
        float q = 0.f, k = 0.f;
        for (int j = 0; j < CS_; ++j) {
            q += Q1[c * CS_ + j] * Watt1[j];
            k += K1[c * CS_ + j] * Watt1[CS_ + j];
        }
        sml[SML_QV1 + c] = q;
        sml[SML_KV1 + c] = k;
    }
    if (c < COUT_) {
        float q = 0.f, k = 0.f;
        for (int j = 0; j < COUT_; ++j) {
            q += Q2[c * COUT_ + j] * Watt2[j];
            k += K2[c * COUT_ + j] * Watt2[COUT_ + j];
        }
        sml[SML_QV2 + c] = q;
        sml[SML_KV2 + c] = k;
    }
    if (c == 63) {
        float s1 = 0.f, s2 = 0.f;
        for (int j = 0; j < CS_; ++j)   s1 += We1[j] * Watt1[2 * CS_ + j];
        for (int j = 0; j < COUT_; ++j) s2 += We2[j] * Watt2[2 * COUT_ + j];
        sml[SML_S31] = s1;
        sml[SML_S32] = s2;
    }
}

// ---------------------------------------------------------------------------
// Tall-skinny row GEMM via v_wmma_f32_16x16x32_f16.
//   Y[gr, :] = concat( X0[b, gather(n), 0:C0], X1[b, n, 0:C1] ) @ W (+ bias)
// gr = b*Ndst + n; W is (C0+C1) x COLS row-major f32.
// One wave per 16-row tile. A tile staged as f16 [16][KPAD] via float4 loads
// + b64 LDS stores; weights staged TRANSPOSED as f16 [COLS][KPAD] so both the
// A and B fragments are two contiguous 8-half runs -> ds_load_b128 pairs.
// Fragment mappings follow the CDNA5 ISA 16-bit A(16x32) / f32 C(16x16)
// VGPR layouts. f32 accumulation.
// ---------------------------------------------------------------------------
template <int C0, int C1, int COLS, bool HAS_BIAS>
__global__ __launch_bounds__(128) void gemm_wmma_kernel(
    const float* __restrict__ X0, const float* __restrict__ X1,
    const int* __restrict__ gather, const float* __restrict__ W,
    const float* __restrict__ bias, float* __restrict__ Y,
    int nRows, int Nsrc, int Ndst) {
    constexpr int KTOT = C0 + C1;                 // multiple of 4
    constexpr int KPAD = (KTOT + 31) & ~31;
    constexpr int KSTEPS = KPAD / 32;
    constexpr int CH = KPAD / 4;                  // float4 chunks per A row

    __shared__ _Float16 sWt[COLS * KPAD];         // transposed: [col][k]
    __shared__ _Float16 sA[4][16 * KPAD];

    // stage weights transposed (f32 -> f16), zero-pad K
    for (int t = threadIdx.x; t < COLS * KPAD; t += 128) {
        int c = t / KPAD, k = t - c * KPAD;
        sWt[t] = (k < KTOT) ? (_Float16)W[k * COLS + c] : (_Float16)0.f;
    }

    const int wave = threadIdx.x >> 5;
    const int lane = threadIdx.x & 31;
    const int half = lane >> 4;
    const int l16  = lane & 15;
    const int tile = blockIdx.x * 4 + wave;
    const int nTiles = nRows >> 4;
    const bool active = tile < nTiles;

    if (active) {
        _Float16* A = sA[wave];
        // vectorized A staging: 4 values per op (chunks never straddle C0|C1)
        for (int g = lane; g < 16 * CH; g += 32) {
            int row = g / CH, k4 = (g - row * CH) * 4;
            int gr = tile * 16 + row;
            int b = gr / Ndst, n = gr - b * Ndst;
            float4 v = make_float4(0.f, 0.f, 0.f, 0.f);
            if (k4 < C0) {
                int sn = gather ? gather[n] : n;
                v = *(const float4*)(X0 + (size_t)(b * Nsrc + sn) * C0 + k4);
            } else if (k4 < KTOT) {
                v = *(const float4*)(X1 + (size_t)(b * Ndst + n) * C1 + (k4 - C0));
            }
            v4h hv;
            hv[0] = (_Float16)v.x; hv[1] = (_Float16)v.y;
            hv[2] = (_Float16)v.z; hv[3] = (_Float16)v.w;
            *(v4h*)(A + row * KPAD + k4) = hv;
        }
    }
    __syncthreads();
    if (!active) return;

    const _Float16* A = sA[wave];
    for (int ct = 0; ct < COLS / 16; ++ct) {
        const int col = ct * 16 + l16;
        v8f acc = {};
        for (int ks = 0; ks < KSTEPS; ++ks) {
            // A frag: lane = row M, k runs half*8..+7 and 16+half*8..+7
            const v8h* ap = (const v8h*)(A + l16 * KPAD + ks * 32 + half * 8);
            v8h alo = ap[0];
            v8h ahi = ap[2];  // +16 halves
            // B frag: lane = col N, identical contiguous k runs (transposed W)
            const v8h* bp = (const v8h*)(sWt + col * KPAD + ks * 32 + half * 8);
            v8h blo = bp[0];
            v8h bhi = bp[2];
            v16h af, bf;
#pragma unroll
            for (int i = 0; i < 8; ++i) {
                af[i] = alo[i]; af[i + 8] = ahi[i];
                bf[i] = blo[i]; bf[i + 8] = bhi[i];
            }
            acc = __builtin_amdgcn_wmma_f32_16x16x32_f16(false, af, false, bf,
                                                         (short)0, acc, false, false);
        }
        const float bv = HAS_BIAS ? bias[col] : 0.f;  // hoisted, branch-free
#pragma unroll
        for (int r = 0; r < 8; ++r) {
            int rl = r + half * 8;           // C layout: VGPR r -> M=r / M=8+r
            int gr = tile * 16 + rl;
            Y[(size_t)gr * COLS + col] = acc[r] + bv;
        }
    }
}

// ---------------------------------------------------------------------------
// Edge phase: att = sigmoid(x_j.qv + x_i.kv + w*s3 + batt)
//             msg_c = att * x_j_c * sigmoid(w*We_c);  aggr[b,dst,c] += msg_c
// One thread per (b, edge); float4 gathers (node tables are L2-resident).
// ---------------------------------------------------------------------------
template <int C>
__global__ __launch_bounds__(256) void edge_kernel(
    const float* __restrict__ xs, const float* __restrict__ xd,
    const int* __restrict__ ei, const float* __restrict__ ew,
    const float* __restrict__ sml, int qoff, int koff, int s3off,
    const float* __restrict__ batt, const float* __restrict__ We,
    float* __restrict__ aggr, int E, int Nsrc, int Ndst) {
    constexpr int C4 = C / 4;
    int t = blockIdx.x * 256 + threadIdx.x;
    if (t >= BB * E) return;
    int b = t / E;
    int e = t - b * E;
    int src = ei[e];
    int dst = ei[E + e];
    float w = ew[e];

    const float4* xj4 = (const float4*)(xs + (size_t)(b * Nsrc + src) * C);
    const float4* xi4 = (const float4*)(xd + (size_t)(b * Ndst + dst) * C);
    const float4* qv4 = (const float4*)(sml + qoff);
    const float4* kv4 = (const float4*)(sml + koff);

    float4 xj[C4];
    float dq = 0.f, dk = 0.f;
#pragma unroll
    for (int q = 0; q < C4; ++q) {
        float4 a = xj4[q];
        float4 bq = xi4[q];
        float4 qv = qv4[q];
        float4 kv = kv4[q];
        xj[q] = a;
        dq += a.x * qv.x + a.y * qv.y + a.z * qv.z + a.w * qv.w;
        dk += bq.x * kv.x + bq.y * kv.y + bq.z * kv.z + bq.w * kv.w;
    }
    float z = dq + dk + w * sml[s3off] + batt[0];
    float att = 1.f / (1.f + __expf(-z));

    const float4* we4 = (const float4*)We;
    float* ag = aggr + (size_t)(b * Ndst + dst) * C;
#pragma unroll
    for (int q = 0; q < C4; ++q) {
        float4 wv = we4[q];
        float4 a = xj[q];
        atomicAdd(&ag[q * 4 + 0], att * a.x / (1.f + __expf(-w * wv.x)));
        atomicAdd(&ag[q * 4 + 1], att * a.y / (1.f + __expf(-w * wv.y)));
        atomicAdd(&ag[q * 4 + 2], att * a.z / (1.f + __expf(-w * wv.z)));
        atomicAdd(&ag[q * 4 + 3], att * a.w / (1.f + __expf(-w * wv.w)));
    }
}

// ---------------------------------------------------------------------------
// Per-node normalization over (B, C) with ddof=1, residual add, leaky-ReLU.
// ---------------------------------------------------------------------------
template <int C>
__global__ __launch_bounds__(256) void norm_kernel(
    const float* __restrict__ pre, const float* __restrict__ xd,
    float* __restrict__ h, int Ndst) {
    constexpr int C4 = C / 4;
    int n = blockIdx.x * 256 + threadIdx.x;
    if (n >= Ndst) return;
    const float cnt = (float)(BB * C);
    float s = 0.f, s2 = 0.f;
    for (int b = 0; b < BB; ++b) {
        const float4* p = (const float4*)(pre + (size_t)(b * Ndst + n) * C);
#pragma unroll
        for (int q = 0; q < C4; ++q) {
            float4 v = p[q];
            s  += v.x + v.y + v.z + v.w;
            s2 += v.x * v.x + v.y * v.y + v.z * v.z + v.w * v.w;
        }
    }
    float mean = s / cnt;
    float var  = (s2 - cnt * mean * mean) / (cnt - 1.f);
    float rstd = rsqrtf(var + 1e-5f);
    for (int b = 0; b < BB; ++b) {
        const float4* p = (const float4*)(pre + (size_t)(b * Ndst + n) * C);
        const float4* x = (const float4*)(xd + (size_t)(b * Ndst + n) * C);
        float4* o = (float4*)(h + (size_t)(b * Ndst + n) * C);
#pragma unroll
        for (int q = 0; q < C4; ++q) {
            float4 pv = p[q];
            float4 xv = x[q];
            float4 r;
            r.x = xv.x + (pv.x - mean) * rstd;
            r.y = xv.y + (pv.y - mean) * rstd;
            r.z = xv.z + (pv.z - mean) * rstd;
            r.w = xv.w + (pv.w - mean) * rstd;
            r.x = r.x > 0.f ? r.x : 0.01f * r.x;
            r.y = r.y > 0.f ? r.y : 0.01f * r.y;
            r.z = r.z > 0.f ? r.z : 0.01f * r.z;
            r.w = r.w > 0.f ? r.w : 0.01f * r.w;
            o[q] = r;
        }
    }
}

// ---------------------------------------------------------------------------
// Launcher
// ---------------------------------------------------------------------------
extern "C" void kernel_launch(void* const* d_in, const int* in_sizes, int n_in,
                              void* d_out, int out_size, void* d_ws, size_t ws_size,
                              hipStream_t stream) {
    (void)in_sizes; (void)n_in; (void)out_size; (void)ws_size;
    const float* X     = (const float*)d_in[0];
    const int*   ei0   = (const int*)d_in[1];
    const float* ew0   = (const float*)d_in[2];
    const int*   rid0  = (const int*)d_in[3];
    const int*   ei1   = (const int*)d_in[4];
    const float* ew1   = (const float*)d_in[5];
    const int*   rid1  = (const int*)d_in[6];
    const float* Wn1   = (const float*)d_in[7];
    const float* We1   = (const float*)d_in[8];
    const float* Q1    = (const float*)d_in[9];
    const float* K1    = (const float*)d_in[10];
    const float* Watt1 = (const float*)d_in[11];
    const float* batt1 = (const float*)d_in[12];
    const float* Wc1   = (const float*)d_in[13];
    const float* bc1   = (const float*)d_in[14];
    const float* Wn2   = (const float*)d_in[15];
    const float* We2   = (const float*)d_in[16];
    const float* Q2    = (const float*)d_in[17];
    const float* K2    = (const float*)d_in[18];
    const float* Watt2 = (const float*)d_in[19];
    const float* batt2 = (const float*)d_in[20];
    const float* Wc2   = (const float*)d_in[21];
    const float* bc2   = (const float*)d_in[22];

    // workspace layout (floats)
    float* ws   = (float*)d_ws;
    float* sml  = ws;
    float* xs1  = sml  + SML_SIZE;
    float* xd1  = xs1  + (size_t)BB * N0_ * CS_;
    float* ag1  = xd1  + (size_t)BB * N1_ * CS_;
    float* pre1 = ag1  + (size_t)BB * N1_ * CS_;
    float* h1   = pre1 + (size_t)BB * N1_ * CS_;
    float* xs2  = h1   + (size_t)BB * N1_ * CS_;
    float* xd2  = xs2  + (size_t)BB * N1_ * COUT_;
    float* ag2  = xd2  + (size_t)BB * N2_ * COUT_;
    float* pre2 = ag2  + (size_t)BB * N2_ * COUT_;
    float* out  = (float*)d_out;

    // zero the scatter targets (deterministic each call; capture-safe)
    hipMemsetAsync(ag1, 0, (size_t)BB * N1_ * CS_ * sizeof(float), stream);
    hipMemsetAsync(ag2, 0, (size_t)BB * N2_ * COUT_ * sizeof(float), stream);

    prep_kernel<<<1, 64, 0, stream>>>(Q1, K1, Watt1, We1, Q2, K2, Watt2, We2, sml);

    auto blocksFor = [](int nRows) { return ((nRows >> 4) + 3) / 4; };

    // ---- layer 1 ----
    gemm_wmma_kernel<CIN_, 0, CS_, false><<<blocksFor(BB * N0_), 128, 0, stream>>>(
        X, nullptr, nullptr, Wn1, nullptr, xs1, BB * N0_, N0_, N0_);
    gemm_wmma_kernel<CIN_, 0, CS_, false><<<blocksFor(BB * N1_), 128, 0, stream>>>(
        X, nullptr, rid0, Wn1, nullptr, xd1, BB * N1_, N0_, N1_);
    edge_kernel<CS_><<<(BB * E0_ + 255) / 256, 256, 0, stream>>>(
        xs1, xd1, ei0, ew0, sml, SML_QV1, SML_KV1, SML_S31, batt1, We1,
        ag1, E0_, N0_, N1_);
    gemm_wmma_kernel<CS_, CS_, CS_, true><<<blocksFor(BB * N1_), 128, 0, stream>>>(
        xd1, ag1, nullptr, Wc1, bc1, pre1, BB * N1_, N1_, N1_);
    norm_kernel<CS_><<<(N1_ + 255) / 256, 256, 0, stream>>>(pre1, xd1, h1, N1_);

    // ---- layer 2 ----
    gemm_wmma_kernel<CS_, 0, COUT_, false><<<blocksFor(BB * N1_), 128, 0, stream>>>(
        h1, nullptr, nullptr, Wn2, nullptr, xs2, BB * N1_, N1_, N1_);
    gemm_wmma_kernel<CS_, 0, COUT_, false><<<blocksFor(BB * N2_), 128, 0, stream>>>(
        h1, nullptr, rid1, Wn2, nullptr, xd2, BB * N2_, N1_, N2_);
    edge_kernel<COUT_><<<(BB * E1_ + 255) / 256, 256, 0, stream>>>(
        xs2, xd2, ei1, ew1, sml, SML_QV2, SML_KV2, SML_S32, batt2, We2,
        ag2, E1_, N1_, N2_);
    gemm_wmma_kernel<COUT_, COUT_, COUT_, true><<<blocksFor(BB * N2_), 128, 0, stream>>>(
        xd2, ag2, nullptr, Wc2, bc2, pre2, BB * N2_, N2_, N2_);
    norm_kernel<COUT_><<<(N2_ + 255) / 256, 256, 0, stream>>>(pre2, xd2, out, N2_);
}
